// Processor_29137058136348
// MI455X (gfx1250) — matrix operand order repaired and checked by reading, not compiled, not run on hardware
//
#include <hip/hip_runtime.h>

// ---------------------------------------------------------------------------
// MeshGraphNet processor for MI455X (gfx1250, wave32, WMMA, TDM).
// Per-MLP fused kernel: gather/concat -> bf16 GEMM1(384->128) -> ReLU ->
// bf16 GEMM2(128->128) -> LayerNorm -> residual. Weights are pre-converted
// once into a packed transposed bf16 image (L2-resident), then DMA'd into
// LDS by the Tensor Data Mover (tensor_load_to_lds + s_wait_tensorcnt).
// Scatter-mean via f32 global atomics.
// ---------------------------------------------------------------------------

#define FL 128          // feature length L
#define FK 384          // concat input 3*L
#define TILE_R 64       // rows per block
#define WPACK 65536     // ushorts per packed weight slot: W1t(128x384)+W2t(128x128)

typedef __bf16 bf16_t;
typedef bf16_t v16bf __attribute__((ext_vector_type(16)));
typedef float  v8f   __attribute__((ext_vector_type(8)));
typedef unsigned int v4u __attribute__((ext_vector_type(4)));
typedef int          v4i __attribute__((ext_vector_type(4)));
typedef int          v8i __attribute__((ext_vector_type(8)));

#if defined(__has_builtin)
#if __has_builtin(__builtin_amdgcn_tensor_load_to_lds) && __has_builtin(__builtin_amdgcn_s_wait_tensorcnt)
#define HAVE_TDM 1
#endif
#endif
#ifndef HAVE_TDM
#define HAVE_TDM 0
#endif

__device__ inline unsigned short f2bf(float f) {
  unsigned int u = __float_as_uint(f);
  unsigned int r = 0x7FFFu + ((u >> 16) & 1u);   // round-to-nearest-even
  return (unsigned short)((u + r) >> 16);
}

union FragBF { v16bf v; unsigned int u[8]; uint4 q[2]; };

// A fragment (16x32 bf16), row-major LDS [row][k]. For half-lane group hl:
// u[0..3] = K = kk+8*hl .. +7 (16 contiguous bytes), u[4..7] = K = kk+16+8*hl.
// Two ds_load_b128 per fragment.
__device__ inline v16bf load_a(const unsigned short* base, int row, int stride,
                               int kk, int hl) {
  const unsigned short* p = base + row * stride + kk + 8 * hl;
  FragBF f;
  f.q[0] = *(const uint4*)(p);
  f.q[1] = *(const uint4*)(p + 16);
  return f.v;
}

// B fragment (32x16 bf16) from transposed weights [n][k]: u[0..3] = K =
// kk+8*hl .. +7, u[4..7] = K = kk+16+8*hl. Two ds_load_b128 per fragment.
__device__ inline v16bf load_b(const unsigned short* baseT, int n, int strideK,
                               int kk, int hl) {
  const unsigned short* p = baseT + n * strideK + kk + 8 * hl;
  FragBF f;
  f.q[0] = *(const uint4*)(p);
  f.q[1] = *(const uint4*)(p + 16);
  return f.v;
}

// ---- One-time weight pre-pack: f32 [k][n] -> bf16 transposed [n][k] -------
// Slot layout (WPACK ushorts): [0,49152) = W1t 128x384 ; [49152,65536) = W2t 128x128
__global__ void prep_weights_kernel(const float* __restrict__ W1,
                                    const float* __restrict__ W2,
                                    unsigned short* __restrict__ out, int S) {
  long i = (long)blockIdx.x * blockDim.x + threadIdx.x;
  int s = (int)(i >> 16);
  int off = (int)(i & (WPACK - 1));
  if (s >= S) return;
  unsigned short v;
  if (off < FK * FL) {
    int n = off / FK, k = off % FK;
    v = f2bf(W1[(size_t)s * FK * FL + (size_t)k * FL + n]);
  } else {
    int o = off - FK * FL;
    int n = o / FL, k = o % FL;
    v = f2bf(W2[(size_t)s * FL * FL + (size_t)k * FL + n]);
  }
  out[i] = v;
}

// mode 0 (edge): Xc = [x[idx[row]], x[idx[E+row]], aA[row]]; dst==aA residual.
// mode 1 (node): Xc = [x[row], aA[row]/max(cA,1), aB[row]/max(cB,1)]; dst==x.
__global__ void __launch_bounds__(256)
mlp_fused_kernel(const float* __restrict__ x,
                 const float* __restrict__ aA,
                 const float* __restrict__ aB,
                 const float* __restrict__ cA,
                 const float* __restrict__ cB,
                 const int*   __restrict__ idx,
                 float* __restrict__ dst,
                 int R, int E, int mode,
                 const unsigned short* __restrict__ Wpk,   // packed bf16 W1t|W2t
                 const float* __restrict__ b1, const float* __restrict__ b2,
                 const float* __restrict__ g,  const float* __restrict__ beta) {
  extern __shared__ char smem[];
  unsigned short* sW1t = (unsigned short*)(smem);           // 128*384*2 = 98304
  unsigned short* sW2t = (unsigned short*)(smem + 98304);   // 128*128*2 = 32768
  float* sB1   = (float*)(smem + 131072);                   // 4*128*4   = 2048
  float* sB2   = sB1 + FL;
  float* sG    = sB2 + FL;
  float* sBeta = sG + FL;
  unsigned short* sXc = (unsigned short*)(smem + 133120);   // 64*384*2  = 49152
  unsigned short* sH1 = (unsigned short*)(smem + 182272);   // 64*128*2  = 16384
  float* sH2 = (float*)(smem + 198656);                     // 64*128*4  = 32768
                                                            // total 231424 B
  const int tid  = threadIdx.x;
  const int lane = tid & 31;
  const int wave = tid >> 5;
  const int hl   = lane >> 4;
  const int base = blockIdx.x * TILE_R;

  __builtin_prefetch(Wpk + (size_t)tid * 256, 0, 0);   // global_prefetch_b8

  // ---- Stage packed weights (131072 B, contiguous) into LDS ---------------
#if HAVE_TDM
  if (wave == 0) {
    // Tensor DMA descriptor: 1D tile, 32768 dwords (data_size=4B), type=image.
    unsigned long long ga = (unsigned long long)Wpk;
    unsigned lds_off = (unsigned)(unsigned long long)(void*)sW1t; // addr[31:0] = LDS offset
    v4u g0;
    g0[0] = 1u;                                   // count=1, user descriptor
    g0[1] = lds_off;                              // lds_addr
    g0[2] = (unsigned)(ga & 0xFFFFFFFFu);         // global_addr[31:0]
    g0[3] = ((unsigned)(ga >> 32) & 0x01FFFFFFu)  // global_addr[56:32]
            | 0x80000000u;                        // type=2 ("image")
    v8i g1;
    g1[0] = 0x20000;                              // data_size=2 (4 bytes), no mask
    g1[1] = (int)(32768u << 16);                  // tensor_dim0 = 32768 (lo16 in [31:16])
    g1[2] = (int)(1u << 16);                      // tensor_dim0 hi=0, tensor_dim1 = 1
    g1[3] = (int)(32768u << 16);                  // tile_dim0 = 32768
    g1[4] = 1;                                    // tile_dim1 = 1, tile_dim2 = 0
    g1[5] = 32768;                                // tensor_dim0_stride = 32768
    g1[6] = 0;
    g1[7] = 0;
    v4i z4 = {0, 0, 0, 0};
#if __clang_major__ >= 23
    v8i z8 = {0, 0, 0, 0, 0, 0, 0, 0};
    __builtin_amdgcn_tensor_load_to_lds(g0, g1, z4, z4, z8, 0);
#else
    __builtin_amdgcn_tensor_load_to_lds(g0, g1, z4, z4, 0);
#endif
    __builtin_amdgcn_s_wait_tensorcnt(0);
  }
#else
  {
    const uint4* src = (const uint4*)Wpk;
    uint4* dl = (uint4*)sW1t;                     // sW1t|sW2t are contiguous
    for (int t = tid; t < (WPACK * 2) / 16; t += 256) dl[t] = src[t];
  }
#endif
  for (int t = tid; t < FL; t += 256) {
    sB1[t] = b1[t]; sB2[t] = b2[t]; sG[t] = g[t]; sBeta[t] = beta[t];
  }

  // ---- Build concatenated input tile (TILE_R x 384) in bf16 ---------------
  for (int t = tid; t < TILE_R * (FK / 4); t += 256) {
    int r = t / (FK / 4);
    int c = (t % (FK / 4)) * 4;
    int row = base + r;
    float4 val = make_float4(0.f, 0.f, 0.f, 0.f);
    if (row < R) {
      if (mode == 0) {
        if (c < FL)            { int s0 = idx[row];
                                 val = *(const float4*)(x + (size_t)s0 * FL + c); }
        else if (c < 2 * FL)   { int s1 = idx[E + row];
                                 val = *(const float4*)(x + (size_t)s1 * FL + (c - FL)); }
        else                   { val = *(const float4*)(aA + (size_t)row * FL + (c - 2 * FL)); }
      } else {
        if (c < FL)            { val = *(const float4*)(x + (size_t)row * FL + c); }
        else if (c < 2 * FL)   { float inv = 1.f / fmaxf(cA[row], 1.f);
                                 float4 v4 = *(const float4*)(aA + (size_t)row * FL + (c - FL));
                                 val = make_float4(v4.x*inv, v4.y*inv, v4.z*inv, v4.w*inv); }
        else                   { float inv = 1.f / fmaxf(cB[row], 1.f);
                                 float4 v4 = *(const float4*)(aB + (size_t)row * FL + (c - 2 * FL));
                                 val = make_float4(v4.x*inv, v4.y*inv, v4.z*inv, v4.w*inv); }
      }
    }
    unsigned int p0 = (unsigned int)f2bf(val.x) | ((unsigned int)f2bf(val.y) << 16);
    unsigned int p1 = (unsigned int)f2bf(val.z) | ((unsigned int)f2bf(val.w) << 16);
    *(unsigned int*)(sXc + r * FK + c)     = p0;
    *(unsigned int*)(sXc + r * FK + c + 2) = p1;
  }
  __syncthreads();

  // ---- GEMM1: H1 = relu(Xc @ W1 + b1)   (64x384)x(384x128) ---------------
  const int nn = wave * 16 + (lane & 15);
  v8f acc[4];
#pragma unroll
  for (int mi = 0; mi < 4; ++mi)
#pragma unroll
    for (int i = 0; i < 8; ++i) acc[mi][i] = 0.f;

  for (int kk = 0; kk < FK; kk += 32) {
    v16bf bf = load_b(sW1t, nn, FK, kk, hl);
#pragma unroll
    for (int mi = 0; mi < 4; ++mi) {
      v16bf af = load_a(sXc, mi * 16 + (lane & 15), FK, kk, hl);
      acc[mi] = __builtin_amdgcn_wmma_f32_16x16x32_bf16(
          false, af, false, bf, (short)0, acc[mi], false, false);
    }
  }
  {
    float bias = sB1[nn];
#pragma unroll
    for (int mi = 0; mi < 4; ++mi)
#pragma unroll
      for (int i = 0; i < 8; ++i) {
        float h = acc[mi][i] + bias;      // C/D layout: VGPR i -> M = i + 8*hl
        h = h > 0.f ? h : 0.f;
        sH1[(mi * 16 + i + 8 * hl) * FL + nn] = f2bf(h);
      }
  }
  __syncthreads();

  // ---- GEMM2: H2 = H1 @ W2 + b2   (64x128)x(128x128) ---------------------
#pragma unroll
  for (int mi = 0; mi < 4; ++mi)
#pragma unroll
    for (int i = 0; i < 8; ++i) acc[mi][i] = 0.f;
  for (int kk = 0; kk < FL; kk += 32) {
    v16bf bf = load_b(sW2t, nn, FL, kk, hl);
#pragma unroll
    for (int mi = 0; mi < 4; ++mi) {
      v16bf af = load_a(sH1, mi * 16 + (lane & 15), FL, kk, hl);
      acc[mi] = __builtin_amdgcn_wmma_f32_16x16x32_bf16(
          false, af, false, bf, (short)0, acc[mi], false, false);
    }
  }
  {
    float bias = sB2[nn];
#pragma unroll
    for (int mi = 0; mi < 4; ++mi)
#pragma unroll
      for (int i = 0; i < 8; ++i)
        sH2[(mi * 16 + i + 8 * hl) * FL + nn] = acc[mi][i] + bias;
  }
  __syncthreads();

  // ---- LayerNorm + residual: 4 lanes/row, wave32 shfl reduction -----------
  {
    int row_l = tid >> 2, sub = tid & 3;
    int row = base + row_l;
    const float* h = sH2 + row_l * FL;
    int c0 = sub * 32;
    float lsum = 0.f, lsq = 0.f;
#pragma unroll
    for (int c = 0; c < 32; ++c) { float v = h[c0 + c]; lsum += v; lsq += v * v; }
    lsum += __shfl_xor(lsum, 1, 32); lsq += __shfl_xor(lsq, 1, 32);
    lsum += __shfl_xor(lsum, 2, 32); lsq += __shfl_xor(lsq, 2, 32);
    float mu  = lsum * (1.f / FL);
    float var = lsq * (1.f / FL) - mu * mu;
    float rinv = rsqrtf(var + 1e-5f);
    if (row < R) {
      float4* dp = (float4*)(dst + (size_t)row * FL + c0);
      const float4* hp = (const float4*)(h + c0);
      const float4* gp = (const float4*)(sG + c0);
      const float4* bp = (const float4*)(sBeta + c0);
#pragma unroll
      for (int q = 0; q < 8; ++q) {
        float4 hv = hp[q], gv = gp[q], bv = bp[q], dv = dp[q];
        dv.x += gv.x * (hv.x - mu) * rinv + bv.x;
        dv.y += gv.y * (hv.y - mu) * rinv + bv.y;
        dv.z += gv.z * (hv.z - mu) * rinv + bv.z;
        dv.w += gv.w * (hv.w - mu) * rinv + bv.w;
        dp[q] = dv;
      }
    }
  }
}

__global__ void zero_kernel(float* __restrict__ p, long n) {
  long i = (long)blockIdx.x * blockDim.x + threadIdx.x;
  if (i < n) p[i] = 0.f;
}

// 32 lanes per edge, 4 contiguous floats per lane -> global_atomic_add_f32.
__global__ void scatter_add_kernel(const float* __restrict__ attr,
                                   const int* __restrict__ recv,
                                   float* __restrict__ sum,
                                   float* __restrict__ cnt, int E) {
  int t = blockIdx.x * blockDim.x + threadIdx.x;
  int e = t >> 5;
  int ln = t & 31;
  if (e >= E) return;
  int r = recv[e];
  const float* a = attr + (size_t)e * FL + ln * 4;
  float* s = sum + (size_t)r * FL + ln * 4;
  atomicAdd(s + 0, a[0]);
  atomicAdd(s + 1, a[1]);
  atomicAdd(s + 2, a[2]);
  atomicAdd(s + 3, a[3]);
  if (ln == 0) atomicAdd(cnt + r, 1.f);
}

extern "C" void kernel_launch(void* const* d_in, const int* in_sizes, int n_in,
                              void* d_out, int out_size, void* d_ws, size_t ws_size,
                              hipStream_t stream) {
  const int Lc = FL;
  const int N  = in_sizes[0] / Lc;
  const int EM = in_sizes[1] / Lc;
  const int EW = in_sizes[2] / Lc;
  const int S  = in_sizes[6] / Lc;   // me_b1 is (S, L)

  const int* midx = (const int*)d_in[3];
  const int* widx = (const int*)d_in[4];
  const float* P[18];
  for (int i = 0; i < 18; ++i) P[i] = (const float*)d_in[5 + i];
  // P[0..5]=me{W1,b1,W2,b2,g,beta}, P[6..11]=we{...}, P[12..17]=nd{...}

  // Working copies live in d_out (x | mea | wea), updated in place per step.
  float* xb  = (float*)d_out;
  float* meb = xb  + (size_t)N  * Lc;
  float* web = meb + (size_t)EM * Lc;
  hipMemcpyAsync(xb,  d_in[0], (size_t)N  * Lc * sizeof(float), hipMemcpyDeviceToDevice, stream);
  hipMemcpyAsync(meb, d_in[1], (size_t)EM * Lc * sizeof(float), hipMemcpyDeviceToDevice, stream);
  hipMemcpyAsync(web, d_in[2], (size_t)EW * Lc * sizeof(float), hipMemcpyDeviceToDevice, stream);

  // Workspace: scatter-mean accumulators + packed bf16 weights.
  float* summ = (float*)d_ws;
  float* sumw = summ + (size_t)N * Lc;
  float* cntm = sumw + (size_t)N * Lc;
  float* cntw = cntm + N;
  long zn = (long)N * Lc * 2 + 2L * N;
  unsigned short* wpack = (unsigned short*)(cntw + N);  // 3*S*WPACK ushorts

  // One-time (per call) weight pre-pack: f32 -> transposed bf16, L2-resident.
  {
    int blocks = (int)(((long)S * WPACK + 255) / 256);
    prep_weights_kernel<<<blocks, 256, 0, stream>>>(P[0],  P[2],  wpack,                      S);
    prep_weights_kernel<<<blocks, 256, 0, stream>>>(P[6],  P[8],  wpack + (size_t)S * WPACK,  S);
    prep_weights_kernel<<<blocks, 256, 0, stream>>>(P[12], P[14], wpack + (size_t)2 * S * WPACK, S);
  }

  const size_t SMEM = 231424;            // 226 KB of the 320 KB/WGP LDS
  dim3 blk(256);
  const size_t vs = Lc;

  for (int s = 0; s < S; ++s) {
    const unsigned short* mW = wpack + (size_t)s * WPACK;
    const unsigned short* wW = wpack + (size_t)(S + s) * WPACK;
    const unsigned short* nW = wpack + (size_t)(2 * S + s) * WPACK;
    // mesh edge MLP (residual into mea)
    mlp_fused_kernel<<<(EM + TILE_R - 1) / TILE_R, blk, SMEM, stream>>>(
        xb, meb, nullptr, nullptr, nullptr, midx, meb, EM, EM, 0,
        mW, P[1] + s * vs, P[3] + s * vs, P[4] + s * vs, P[5] + s * vs);
    // world edge MLP (residual into wea)
    mlp_fused_kernel<<<(EW + TILE_R - 1) / TILE_R, blk, SMEM, stream>>>(
        xb, web, nullptr, nullptr, nullptr, widx, web, EW, EW, 0,
        wW, P[7] + s * vs, P[9] + s * vs, P[10] + s * vs, P[11] + s * vs);
    // scatter-mean accumulators
    zero_kernel<<<(int)((zn + 255) / 256), blk, 0, stream>>>(summ, zn);
    scatter_add_kernel<<<(EM * 32 + 255) / 256, blk, 0, stream>>>(meb, midx + EM, summ, cntm, EM);
    scatter_add_kernel<<<(EW * 32 + 255) / 256, blk, 0, stream>>>(web, widx + EW, sumw, cntw, EW);
    // node MLP (residual into x)
    mlp_fused_kernel<<<(N + TILE_R - 1) / TILE_R, blk, SMEM, stream>>>(
        xb, summ, sumw, cntm, cntw, nullptr, xb, N, 0, 1,
        nW, P[13] + s * vs, P[15] + s * vs, P[16] + s * vs, P[17] + s * vs);
  }
}
